// CGC_Layer_29205777613114
// MI455X (gfx1250) — matrix operand order repaired and checked by reading, not compiled, not run on hardware
//
#include <hip/hip_runtime.h>

typedef unsigned short u16;
typedef __attribute__((ext_vector_type(16))) __bf16 v16bf;
typedef __attribute__((ext_vector_type(8)))  float  v8f;

#define B_  8192
#define D_  512
#define H_  1024
#define O_  512
#define T_  2
#define NE_ 4
#define NS_ 4
#define E_  12          // 8 task experts (t*NE+i) then 4 shared
#define GT_ 8           // NE+NS  task-gate width
#define GS_ 12          // T*NE+NS shared-gate width
#define BM_ 64          // rows per block (4 row-tiles)

// ---------- bf16 helpers (no dependence on __bf16 arithmetic) ----------
static __device__ __forceinline__ float bf2f(u16 h) {
  union { unsigned u; float f; } c; c.u = ((unsigned)h) << 16; return c.f;
}
static __device__ __forceinline__ u16 f2bf(float f) {
  union { float f; unsigned u; } c; c.f = f;
  unsigned u = c.u;
  return (u16)((u + 0x7fffu + ((u >> 16) & 1u)) >> 16);   // RNE
}

static __device__ __forceinline__ v8f wmma_bf16(v16bf a, v16bf b, v8f c) {
  return __builtin_amdgcn_wmma_f32_16x16x32_bf16(false, a, false, b, (short)0, c,
                                                 false, false);
}

// ---------------------------------------------------------------------
// Zero the output (harness poisons d_out; we accumulate with atomics).
// ---------------------------------------------------------------------
__global__ void k_zero(float* p, int n) {
  for (int i = blockIdx.x * blockDim.x + threadIdx.x; i < n;
       i += gridDim.x * blockDim.x)
    p[i] = 0.f;
}

// ---------------------------------------------------------------------
// Pack activations fp32 -> bf16 in WMMA A-fragment order.
// A 16x32 bf16 tile: lane L -> row m=L%16, half=L/16; element i=2v+p ->
// k = (v/4)*16 + half*8 + (v%4)*2 + p.   Layout: [rowtile][ktile][lane][16]
// ---------------------------------------------------------------------
__global__ void k_pack_x(const float* __restrict__ shared_input,
                         const float* __restrict__ task_x,
                         u16* __restrict__ xfS, u16* __restrict__ xfT) {
  const int N = (1 + T_) * B_ * D_;
  const int KT = D_ / 32;
  for (int idx = blockIdx.x * blockDim.x + threadIdx.x; idx < N;
       idx += gridDim.x * blockDim.x) {
    int q = idx / (B_ * D_);
    int f = idx % (B_ * D_);
    int i    = f & 15;
    int lane = (f >> 4) & 31;
    int tile = f >> 9;
    int kt = tile % KT;
    int rt = tile / KT;
    int m = lane & 15, half = lane >> 4, v = i >> 1, p = i & 1;
    int k = kt * 32 + (v >> 2) * 16 + half * 8 + (v & 3) * 2 + p;
    int row = rt * 16 + m;
    float val = (q == 0) ? shared_input[row * D_ + k]
                         : task_x[(q - 1) * (B_ * D_) + row * D_ + k];
    u16 o = f2bf(val);
    if (q == 0) xfS[f] = o;
    else        xfT[(q - 1) * (B_ * D_) + f] = o;
  }
}

// ---------------------------------------------------------------------
// Pack weights fp32 -> bf16 in WMMA B-fragment order.
// B 32x16 bf16 tile: lane L -> col n=L%16; element i -> k = (i/8)*16 + (L/16)*8 + i%8.
// W1 layout: [e][kt=16][nt=64][lane][16] ; W2: [e][kt=32][nt=32][lane][16]
// ---------------------------------------------------------------------
__global__ void k_pack_w(const float* __restrict__ sW1, const float* __restrict__ sW2,
                         const float* __restrict__ tW1, const float* __restrict__ tW2,
                         u16* __restrict__ w1f, u16* __restrict__ w2f) {
  const long N1 = (long)E_ * D_ * H_;
  const long N2 = (long)E_ * H_ * O_;
  for (long idx = blockIdx.x * (long)blockDim.x + threadIdx.x; idx < N1 + N2;
       idx += gridDim.x * (long)blockDim.x) {
    if (idx < N1) {
      int e = (int)(idx / (D_ * H_));
      int f = (int)(idx % (D_ * H_));
      int i = f & 15, lane = (f >> 4) & 31, tile = f >> 9;
      int nt = tile % (H_ / 16);
      int kt = tile / (H_ / 16);
      int k = kt * 32 + (i >> 3) * 16 + (lane >> 4) * 8 + (i & 7);
      int n = nt * 16 + (lane & 15);
      const float* src = (e < 8) ? (tW1 + (long)e * D_ * H_)
                                 : (sW1 + (long)(e - 8) * D_ * H_);
      w1f[idx] = f2bf(src[(long)k * H_ + n]);
    } else {
      long j = idx - N1;
      int e = (int)(j / (H_ * O_));
      int f = (int)(j % (H_ * O_));
      int i = f & 15, lane = (f >> 4) & 31, tile = f >> 9;
      int nt = tile % (O_ / 16);
      int kt = tile / (O_ / 16);
      int k = kt * 32 + (i >> 3) * 16 + (lane >> 4) * 8 + (i & 7);
      int n = nt * 16 + (lane & 15);
      const float* src = (e < 8) ? (tW2 + (long)e * H_ * O_)
                                 : (sW2 + (long)(e - 8) * H_ * O_);
      w2f[j] = f2bf(src[(long)k * O_ + n]);
    }
  }
}

// ---------------------------------------------------------------------
// Gate logits + softmax. One thread per row.
// ---------------------------------------------------------------------
__global__ void k_gates(const float* __restrict__ shared_input,
                        const float* __restrict__ task_x,
                        const float* __restrict__ gW, const float* __restrict__ gb,
                        const float* __restrict__ sgW, const float* __restrict__ sgb,
                        float* __restrict__ tgate, float* __restrict__ sgate) {
  int b = blockIdx.x * blockDim.x + threadIdx.x;
  if (b >= B_) return;
  float a0[GT_], a1[GT_], as[GS_];
#pragma unroll
  for (int j = 0; j < GT_; ++j) { a0[j] = 0.f; a1[j] = 0.f; }
#pragma unroll
  for (int j = 0; j < GS_; ++j) as[j] = 0.f;
  for (int d = 0; d < D_; ++d) {
    float xs = shared_input[b * D_ + d];
    float x0 = task_x[b * D_ + d];
    float x1 = task_x[B_ * D_ + b * D_ + d];
#pragma unroll
    for (int j = 0; j < GT_; ++j) {
      a0[j] += x0 * gW[d * GT_ + j];
      a1[j] += x1 * gW[(D_ + d) * GT_ + j];
    }
#pragma unroll
    for (int j = 0; j < GS_; ++j) as[j] += xs * sgW[d * GS_ + j];
  }
  {
    float mx = -1e30f, sum = 0.f;
#pragma unroll
    for (int j = 0; j < GT_; ++j) { a0[j] += gb[j]; mx = fmaxf(mx, a0[j]); }
#pragma unroll
    for (int j = 0; j < GT_; ++j) { a0[j] = __expf(a0[j] - mx); sum += a0[j]; }
    float inv = 1.f / sum;
#pragma unroll
    for (int j = 0; j < GT_; ++j) tgate[b * GT_ + j] = a0[j] * inv;
  }
  {
    float mx = -1e30f, sum = 0.f;
#pragma unroll
    for (int j = 0; j < GT_; ++j) { a1[j] += gb[GT_ + j]; mx = fmaxf(mx, a1[j]); }
#pragma unroll
    for (int j = 0; j < GT_; ++j) { a1[j] = __expf(a1[j] - mx); sum += a1[j]; }
    float inv = 1.f / sum;
#pragma unroll
    for (int j = 0; j < GT_; ++j) tgate[(B_ + b) * GT_ + j] = a1[j] * inv;
  }
  {
    float mx = -1e30f, sum = 0.f;
#pragma unroll
    for (int j = 0; j < GS_; ++j) { as[j] += sgb[j]; mx = fmaxf(mx, as[j]); }
#pragma unroll
    for (int j = 0; j < GS_; ++j) { as[j] = __expf(as[j] - mx); sum += as[j]; }
    float inv = 1.f / sum;
#pragma unroll
    for (int j = 0; j < GS_; ++j) sgate[b * GS_ + j] = as[j] * inv;
  }
}

// ---------------------------------------------------------------------
// Fused expert kernel. grid = (B/64, 12 experts), block = 512 (16 waves).
// Phase 1: GEMM1 [64,512]@[512,1024] in bf16 WMMA, +bias in regs.
// Phase 2: row stats via shfl_xor tree + ds_add_f32; LN+ReLU applied ONCE
//          to own accumulators; written to LDS already in A-fragment order.
// Phase 3: GEMM2 [64,1024]@[1024,512]: pure ds_load_b128 + global_load_b128
//          + v_wmma loop (no VALU in the K-loop).
// Phase 4: +bias, gated global_atomic_add_f32 into the 3 outputs.
// ---------------------------------------------------------------------
__global__ void __launch_bounds__(512, 1)
k_expert(const u16* __restrict__ xfS, const u16* __restrict__ xfT,
         const u16* __restrict__ w1f, const u16* __restrict__ w2f,
         const float* __restrict__ tgate, const float* __restrict__ sgate,
         const float* __restrict__ sb1, const float* __restrict__ sg,
         const float* __restrict__ sbeta, const float* __restrict__ sb2,
         const float* __restrict__ tb1, const float* __restrict__ tg,
         const float* __restrict__ tbeta, const float* __restrict__ tb2,
         float* __restrict__ out) {
  extern __shared__ char smem[];
  u16*   afrag = (u16*)smem;                     // [4 rowtiles][32 kt][32 lane][16]
  float* ssum  = (float*)(smem + 4 * 32 * 32 * 16 * 2);  // [64]
  float* ssq   = ssum + 64;                               // [64]

  const int e  = blockIdx.y;
  const int rb = blockIdx.x;
  const int tid = threadIdx.x;
  const int w = tid >> 5, lane = tid & 31;
  const int half = lane >> 4, nl = lane & 15;
  const int rtb = (w & 1) * 2;   // wave's 2 row-tiles: rtb, rtb+1 (of 4)
  const int cw  = w >> 1;        // wave's col chunk 0..7
  const bool isTask = (e < 8);
  const int t  = e >> 2;
  const int si = e - 8;

  const u16* xf = isTask ? (xfT + (long)t * B_ * D_) : xfS;
  const float* b1 = isTask ? (tb1 + e * H_) : (sb1 + si * H_);
  const float* gg = isTask ? (tg  + e * H_) : (sg  + si * H_);
  const float* bt = isTask ? (tbeta + e * H_) : (sbeta + si * H_);
  const float* b2 = isTask ? (tb2 + e * O_) : (sb2 + si * O_);
  const u16* w1 = w1f + (long)e * D_ * H_;
  const u16* w2 = w2f + (long)e * H_ * O_;

  if (tid < 64) { ssum[tid] = 0.f; ssq[tid] = 0.f; }
  __syncthreads();

  // ---------------- Phase 1: GEMM1 ----------------
  v8f acc[2][8];
#pragma unroll
  for (int r = 0; r < 2; ++r)
#pragma unroll
    for (int c = 0; c < 8; ++c)
#pragma unroll
      for (int v = 0; v < 8; ++v) acc[r][c][v] = 0.f;

  const int KT1 = D_ / 32;  // 16
  const u16* xA0 = xf + (long)(rb * 4 + rtb) * KT1 * 512 + lane * 16;
  const u16* xA1 = xA0 + (long)KT1 * 512;
  const u16* wB1 = w1 + lane * 16;

  for (int kt = 0; kt < KT1; ++kt) {
    v16bf a0 = *(const v16bf*)(xA0 + kt * 512);
    v16bf a1 = *(const v16bf*)(xA1 + kt * 512);
#pragma unroll
    for (int ct = 0; ct < 8; ++ct) {
      v16bf b = *(const v16bf*)(wB1 + (long)(kt * 64 + cw * 8 + ct) * 512);
      acc[0][ct] = wmma_bf16(a0, b, acc[0][ct]);
      acc[1][ct] = wmma_bf16(a1, b, acc[1][ct]);
    }
  }

  // bias into accumulators (LN stats must include b1)
#pragma unroll
  for (int ct = 0; ct < 8; ++ct) {
    float bias = b1[cw * 128 + ct * 16 + nl];
#pragma unroll
    for (int r = 0; r < 2; ++r)
#pragma unroll
      for (int v = 0; v < 8; ++v) acc[r][ct][v] += bias;
  }

  // ---------------- Phase 2a: stats (shfl tree over the 16-lane N group) ----
#pragma unroll
  for (int r = 0; r < 2; ++r)
#pragma unroll
    for (int v = 0; v < 8; ++v) {
      float s = 0.f, q = 0.f;
#pragma unroll
      for (int ct = 0; ct < 8; ++ct) {
        float x = acc[r][ct][v];
        s += x; q += x * x;
      }
#pragma unroll
      for (int m = 1; m < 16; m <<= 1) {
        s += __shfl_xor(s, m, 32);
        q += __shfl_xor(q, m, 32);
      }
      if (nl == 0) {
        int row = (rtb + r) * 16 + v + 8 * half;
        atomicAdd(&ssum[row], s);
        atomicAdd(&ssq[row], q);
      }
    }
  __syncthreads();

  // ---------------- Phase 2b: LN+ReLU once, write A-fragments to LDS -------
  float mu[2][8], rs[2][8];
  const float inv_h = 1.f / (float)H_;
#pragma unroll
  for (int r = 0; r < 2; ++r)
#pragma unroll
    for (int v = 0; v < 8; ++v) {
      int row = (rtb + r) * 16 + v + 8 * half;
      float m_ = ssum[row] * inv_h;
      mu[r][v] = m_;
      rs[r][v] = __frsqrt_rn(ssq[row] * inv_h - m_ * m_ + 1e-5f);
    }
#pragma unroll
  for (int ct = 0; ct < 8; ++ct) {
    int col = cw * 128 + ct * 16 + nl;
    float gc = gg[col], bc = bt[col];
    int kt2   = cw * 4 + (ct >> 1);
    int i     = (ct & 1) * 8 + (nl & 7);
    int lhalf = nl >> 3;
#pragma unroll
    for (int r = 0; r < 2; ++r) {
      int art = rtb + r;
#pragma unroll
      for (int v = 0; v < 8; ++v) {
        int lp = (v + 8 * half) + 16 * lhalf;
        float y = (acc[r][ct][v] - mu[r][v]) * rs[r][v] * gc + bc;
        afrag[((art * 32 + kt2) * 32 + lp) * 16 + i] = f2bf(fmaxf(y, 0.f));
      }
    }
  }
  __syncthreads();

  // ---------------- Phase 3: GEMM2 (pure loads + WMMA) ----------------
  v8f acc2[2][4];
#pragma unroll
  for (int r = 0; r < 2; ++r)
#pragma unroll
    for (int c = 0; c < 4; ++c)
#pragma unroll
      for (int v = 0; v < 8; ++v) acc2[r][c][v] = 0.f;

  const u16* wB2 = w2 + lane * 16;
  const u16* aF0 = afrag + ((rtb + 0) * 32) * 512 + lane * 16;
  const u16* aF1 = afrag + ((rtb + 1) * 32) * 512 + lane * 16;

  for (int kk = 0; kk < H_ / 32; ++kk) {
    v16bf a0 = *(const v16bf*)(aF0 + kk * 512);
    v16bf a1 = *(const v16bf*)(aF1 + kk * 512);
#pragma unroll
    for (int ct = 0; ct < 4; ++ct) {
      v16bf b = *(const v16bf*)(wB2 + (long)(kk * 32 + cw * 4 + ct) * 512);
      acc2[0][ct] = wmma_bf16(a0, b, acc2[0][ct]);
      acc2[1][ct] = wmma_bf16(a1, b, acc2[1][ct]);
    }
  }

  // ---------------- Phase 4: epilogue ----------------
  float g0[2][8], g1[2][8], g2[2][8];
#pragma unroll
  for (int r = 0; r < 2; ++r)
#pragma unroll
    for (int v = 0; v < 8; ++v) {
      int grow = rb * BM_ + (rtb + r) * 16 + v + 8 * half;
      if (isTask) {
        g0[r][v] = sgate[grow * GS_ + e];
        g1[r][v] = tgate[(t * B_ + grow) * GT_ + (e & 3)];
        g2[r][v] = 0.f;
      } else {
        g0[r][v] = sgate[grow * GS_ + 8 + si];
        g1[r][v] = tgate[grow * GT_ + NE_ + si];
        g2[r][v] = tgate[(B_ + grow) * GT_ + NE_ + si];
      }
    }

#pragma unroll
  for (int ct = 0; ct < 4; ++ct) {
    int col = cw * 64 + ct * 16 + nl;
    float bias = b2[col];
#pragma unroll
    for (int r = 0; r < 2; ++r)
#pragma unroll
      for (int v = 0; v < 8; ++v) {
        int grow = rb * BM_ + (rtb + r) * 16 + v + 8 * half;
        float o = acc2[r][ct][v] + bias;
        unsafeAtomicAdd(out + (size_t)grow * O_ + col, g0[r][v] * o);
        if (isTask) {
          unsafeAtomicAdd(out + (size_t)(1 + t) * B_ * O_ + (size_t)grow * O_ + col,
                          g1[r][v] * o);
        } else {
          unsafeAtomicAdd(out + (size_t)B_ * O_ + (size_t)grow * O_ + col,
                          g1[r][v] * o);
          unsafeAtomicAdd(out + (size_t)2 * B_ * O_ + (size_t)grow * O_ + col,
                          g2[r][v] * o);
        }
      }
  }
}

// ---------------------------------------------------------------------
extern "C" void kernel_launch(void* const* d_in, const int* in_sizes, int n_in,
                              void* d_out, int out_size, void* d_ws, size_t ws_size,
                              hipStream_t stream) {
  (void)in_sizes; (void)n_in; (void)out_size; (void)ws_size;
  const float* shared_input = (const float*)d_in[0];
  const float* task_x = (const float*)d_in[1];
  const float* sW1 = (const float*)d_in[2];
  const float* sb1 = (const float*)d_in[3];
  const float* sg  = (const float*)d_in[4];
  const float* sbe = (const float*)d_in[5];
  const float* sW2 = (const float*)d_in[6];
  const float* sb2 = (const float*)d_in[7];
  const float* tW1 = (const float*)d_in[8];
  const float* tb1 = (const float*)d_in[9];
  const float* tg  = (const float*)d_in[10];
  const float* tbe = (const float*)d_in[11];
  const float* tW2 = (const float*)d_in[12];
  const float* tb2 = (const float*)d_in[13];
  const float* gW  = (const float*)d_in[14];
  const float* gb  = (const float*)d_in[15];
  const float* sgW = (const float*)d_in[16];
  const float* sgb = (const float*)d_in[17];
  float* out = (float*)d_out;

  char* ws = (char*)d_ws;
  u16* xfS = (u16*)(ws + 0);                        //  8 MiB
  u16* xfT = (u16*)(ws + 8388608);                  // 16 MiB
  u16* w1f = (u16*)(ws + 25165824);                 // 12 MiB
  u16* w2f = (u16*)(ws + 37748736);                 // 12 MiB
  float* tgate = (float*)(ws + 50331648);           // 512 KiB
  float* sgate = (float*)(ws + 50855936);           // 384 KiB

  k_zero<<<1024, 256, 0, stream>>>(out, 3 * B_ * O_);
  k_pack_x<<<4096, 256, 0, stream>>>(shared_input, task_x, xfS, xfT);
  k_pack_w<<<4096, 256, 0, stream>>>(sW1, sW2, tW1, tW2, w1f, w2f);
  k_gates<<<B_ / 256, 256, 0, stream>>>(shared_input, task_x, gW, gb, sgW, sgb,
                                        tgate, sgate);
  dim3 grid(B_ / BM_, E_);
  size_t shmem = (size_t)4 * 32 * 32 * 16 * 2 + 128 * sizeof(float); // 128 KiB + 512 B
  k_expert<<<grid, 512, shmem, stream>>>(xfS, xfT, w1f, w2f, tgate, sgate,
                                         sb1, sg, sbe, sb2,
                                         tb1, tg, tbe, tb2, out);
}
